// MultiheadAttention_54451595379220
// MI455X (gfx1250) — compile-verified, matrix operand order checked by reading, not run on hardware
//
#include <hip/hip_runtime.h>
#include <hip/hip_bf16.h>
#include <stdint.h>

typedef __attribute__((ext_vector_type(16))) __bf16 v16bf;
typedef __attribute__((ext_vector_type(8)))  float  v8f;

union FragB { v16bf v; uint4 q[2]; unsigned short u[16]; };

__device__ __forceinline__ unsigned short f2bf(float f) {
  union { float f; unsigned u; } c; c.f = f;
  unsigned r = c.u + 0x7FFFu + ((c.u >> 16) & 1u);
  return (unsigned short)(r >> 16);
}

// Async global->LDS copy of 16 bytes (CDNA5, tracked by ASYNCcnt).
__device__ __forceinline__ void async_copy_b128(const void* gaddr, void* lds_ptr) {
  unsigned lds_off = (unsigned)(uintptr_t)lds_ptr;  // low 32 bits = LDS offset
  asm volatile("global_load_async_to_lds_b128 %0, %1, off"
               :: "v"(lds_off), "v"(gaddr) : "memory");
}
__device__ __forceinline__ void async_wait0() {
  asm volatile("s_wait_asynccnt 0x0" ::: "memory");
}

// ---------------------------------------------------------------------------
// Tiled GEMM: C[M,N] = A[M,K] * B[K,N]; A is f32 or bf16, B (weights) is f32
// converted to bf16 on load, C is bf16 or f32. Block tile 128x128, 8 waves,
// each wave computes 32x64 via 2x4 v_wmma_f32_16x16x32_bf16 fragments.
// ---------------------------------------------------------------------------
template<bool A_IS_F32, bool C_IS_BF16>
__global__ __launch_bounds__(256) void gemm_bf16_wmma(
    const void* __restrict__ Ap, const float* __restrict__ Bp,
    void* __restrict__ Cp, int M, int N, int K)
{
  __shared__ unsigned short As[128][40];  // padded rows (80B, 16B-aligned)
  __shared__ unsigned short Bs[128][40];  // B stored transposed: Bs[n][k]

  const int tid  = threadIdx.x;
  const int lane = tid & 31;
  const int wav  = tid >> 5;
  const int wm   = wav & 3;       // 4 M-subtiles of 32
  const int wn   = wav >> 2;      // 2 N-subtiles of 64
  const int grp  = lane >> 4;
  const int l15  = lane & 15;
  const int m0   = blockIdx.y * 128;
  const int n0   = blockIdx.x * 128;

  v8f acc[2][4];
#pragma unroll
  for (int mi = 0; mi < 2; ++mi)
#pragma unroll
    for (int ni = 0; ni < 4; ++ni)
#pragma unroll
      for (int r = 0; r < 8; ++r) acc[mi][ni][r] = 0.f;

  for (int k0 = 0; k0 < K; k0 += 32) {
    __syncthreads();
    // prefetch next K-tile into caches while we convert/stage this one
    if (k0 + 32 < K) {
      if constexpr (A_IS_F32)
        __builtin_prefetch((const float*)Ap +
            (size_t)(m0 + (tid >> 1)) * K + k0 + 32 + (tid & 1) * 16, 0, 1);
      else
        __builtin_prefetch((const unsigned short*)Ap +
            (size_t)(m0 + (tid >> 1)) * K + k0 + 32 + (tid & 1) * 16, 0, 1);
      __builtin_prefetch(Bp + (size_t)(k0 + 32 + (tid >> 3)) * N + n0 + (tid & 7) * 16, 0, 1);
    }
    // ---- A tile 128x32 -> bf16 LDS ----
    if constexpr (A_IS_F32) {
      const float* A = (const float*)Ap;
#pragma unroll
      for (int p = 0; p < 4; ++p) {
        int lin = p * 256 + tid;            // 1024 float4 slots
        int row = lin >> 3;
        int c4  = (lin & 7) << 2;
        float4 f = *(const float4*)(A + (size_t)(m0 + row) * K + k0 + c4);
        unsigned short* dst = &As[row][c4];
        dst[0] = f2bf(f.x); dst[1] = f2bf(f.y);
        dst[2] = f2bf(f.z); dst[3] = f2bf(f.w);
      }
    } else {
      const unsigned short* A = (const unsigned short*)Ap;
#pragma unroll
      for (int p = 0; p < 4; ++p) {
        int lin = p * 256 + tid;
        int row = lin >> 3;
        int c4  = (lin & 7) << 2;
        *(uint2*)&As[row][c4] =
            *(const uint2*)(A + (size_t)(m0 + row) * K + k0 + c4);
      }
    }
    // ---- B tile 32x128 f32 -> bf16 LDS, transposed to Bs[n][k] ----
#pragma unroll
    for (int p = 0; p < 4; ++p) {
      int lin = p * 256 + tid;              // 1024 float4 slots
      int r  = lin >> 5;                    // k (0..31)
      int c4 = (lin & 31) << 2;             // n
      float4 f = *(const float4*)(Bp + (size_t)(k0 + r) * N + n0 + c4);
      Bs[c4 + 0][r] = f2bf(f.x);
      Bs[c4 + 1][r] = f2bf(f.y);
      Bs[c4 + 2][r] = f2bf(f.z);
      Bs[c4 + 3][r] = f2bf(f.w);
    }
    __syncthreads();

    FragB a[2], b[4];
#pragma unroll
    for (int mi = 0; mi < 2; ++mi) {
      const unsigned short* src = &As[wm * 32 + mi * 16 + l15][grp * 8];
      a[mi].q[0] = *(const uint4*)src;
      a[mi].q[1] = *(const uint4*)(src + 16);
    }
#pragma unroll
    for (int ni = 0; ni < 4; ++ni) {
      const unsigned short* src = &Bs[wn * 64 + ni * 16 + l15][grp * 8];
      b[ni].q[0] = *(const uint4*)src;
      b[ni].q[1] = *(const uint4*)(src + 16);
    }
#pragma unroll
    for (int mi = 0; mi < 2; ++mi)
#pragma unroll
      for (int ni = 0; ni < 4; ++ni)
        acc[mi][ni] = __builtin_amdgcn_wmma_f32_16x16x32_bf16(
            false, a[mi].v, false, b[ni].v, (short)0, acc[mi][ni], false, false);
  }

  // ---- epilogue: C-frag layout M = r + 8*grp, N = l15 ----
#pragma unroll
  for (int mi = 0; mi < 2; ++mi)
#pragma unroll
    for (int ni = 0; ni < 4; ++ni)
#pragma unroll
      for (int r = 0; r < 8; ++r) {
        int gm = m0 + wm * 32 + mi * 16 + grp * 8 + r;
        int gn = n0 + wn * 64 + ni * 16 + l15;
        float val = acc[mi][ni][r];
        if constexpr (C_IS_BF16)
          ((unsigned short*)Cp)[(size_t)gm * N + gn] = f2bf(val);
        else
          ((float*)Cp)[(size_t)gm * N + gn] = val;
      }
}

// ---------------------------------------------------------------------------
// Flash attention, causal. Grid (T/64, B*H). 128 threads = 4 waves; each wave
// owns 16 query rows. hq/hk/hv are bf16 [(t*B+b)*1024 + h*64 + d].
// Q and K tiles use CDNA5 async global->LDS copies (ASYNCcnt).
// ---------------------------------------------------------------------------
__global__ __launch_bounds__(128) void flash_attn_wmma(
    const unsigned short* __restrict__ hq,
    const unsigned short* __restrict__ hk,
    const unsigned short* __restrict__ hv,
    unsigned short* __restrict__ vec)
{
  __shared__ unsigned short Qs[64][72];
  __shared__ unsigned short Ks[64][72];
  __shared__ unsigned short Vst[64][72];    // transposed: Vst[d][key]
  __shared__ unsigned short Ps[4][16][72];  // per-wave P slab

  const int tid  = threadIdx.x;
  const int wav  = tid >> 5;
  const int lane = tid & 31;
  const int grp  = lane >> 4;
  const int l15  = lane & 15;
  const int qt   = blockIdx.x;
  const int bh   = blockIdx.y;
  const int h    = bh & 15;
  const int b    = bh >> 4;
  const int q0   = qt * 64;
  const float scale = 0.125f;               // 1/sqrt(64)

  // async-load Q tile (64x64 bf16, 16B per lane per pass)
#pragma unroll
  for (int p = 0; p < 4; ++p) {
    int lin = p * 128 + tid;                // 512 x 16B slots
    int row = lin >> 3;
    int c8  = (lin & 7) << 3;
    async_copy_b128(hq + ((size_t)(q0 + row) * 8 + b) * 1024 + h * 64 + c8,
                    &Qs[row][c8]);
  }
  async_wait0();
  __syncthreads();

  FragB qa[2];
#pragma unroll
  for (int kk = 0; kk < 2; ++kk) {
    const unsigned short* src = &Qs[wav * 16 + l15][kk * 32 + grp * 8];
    qa[kk].q[0] = *(const uint4*)src;
    qa[kk].q[1] = *(const uint4*)(src + 16);
  }

  float m_i[8], l_i[8];
  v8f o[4];
#pragma unroll
  for (int r = 0; r < 8; ++r) { m_i[r] = -3.0e38f; l_i[r] = 0.f; }
#pragma unroll
  for (int dn = 0; dn < 4; ++dn)
#pragma unroll
    for (int r = 0; r < 8; ++r) o[dn][r] = 0.f;

  for (int kt = 0; kt <= qt; ++kt) {
    __syncthreads();
    // K tile: async copy row-major. V tile: load + transpose into Vst.
#pragma unroll
    for (int p = 0; p < 4; ++p) {
      int lin = p * 128 + tid;
      int row = lin >> 3;
      int c8  = (lin & 7) << 3;
      async_copy_b128(hk + ((size_t)(kt * 64 + row) * 8 + b) * 1024 + h * 64 + c8,
                      &Ks[row][c8]);
    }
#pragma unroll
    for (int p = 0; p < 8; ++p) {
      int lin = p * 128 + tid;
      int row = lin >> 4;
      int c4  = (lin & 15) << 2;
      uint2 raw =
          *(const uint2*)(hv + ((size_t)(kt * 64 + row) * 8 + b) * 1024 + h * 64 + c4);
      Vst[c4 + 0][row] = (unsigned short)(raw.x & 0xffffu);
      Vst[c4 + 1][row] = (unsigned short)(raw.x >> 16);
      Vst[c4 + 2][row] = (unsigned short)(raw.y & 0xffffu);
      Vst[c4 + 3][row] = (unsigned short)(raw.y >> 16);
    }
    async_wait0();
    __syncthreads();

    // S = Q * K^T  (B-frag lanes index key j, elements index d -> Ks rows)
    v8f s[4];
#pragma unroll
    for (int jn = 0; jn < 4; ++jn) {
      v8f c;
#pragma unroll
      for (int r = 0; r < 8; ++r) c[r] = 0.f;
#pragma unroll
      for (int kk = 0; kk < 2; ++kk) {
        FragB kb;
        const unsigned short* src = &Ks[jn * 16 + l15][kk * 32 + grp * 8];
        kb.q[0] = *(const uint4*)src;
        kb.q[1] = *(const uint4*)(src + 16);
        c = __builtin_amdgcn_wmma_f32_16x16x32_bf16(
            false, qa[kk].v, false, kb.v, (short)0, c, false, false);
      }
      s[jn] = c;
    }

    // scale + causal mask + online softmax (row = q0 + wav*16 + grp*8 + r)
    const int qrow_base = q0 + wav * 16 + grp * 8;
    float alpha[8];
#pragma unroll
    for (int r = 0; r < 8; ++r) {
      float rm = -3.0e38f;
#pragma unroll
      for (int jn = 0; jn < 4; ++jn) {
        float x = s[jn][r] * scale;
        if (kt == qt) {
          int col = kt * 64 + jn * 16 + l15;
          if (col > qrow_base + r) x = -3.0e38f;
        }
        s[jn][r] = x;
        rm = fmaxf(rm, x);
      }
#pragma unroll
      for (int off = 1; off < 16; off <<= 1)
        rm = fmaxf(rm, __shfl_xor(rm, off, 32));
      float nm = fmaxf(m_i[r], rm);
      alpha[r] = __expf(m_i[r] - nm);
      m_i[r] = nm;
    }
#pragma unroll
    for (int r = 0; r < 8; ++r) {
      float rs = 0.f;
#pragma unroll
      for (int jn = 0; jn < 4; ++jn) {
        float p = __expf(s[jn][r] - m_i[r]);
        rs += p;
        Ps[wav][grp * 8 + r][jn * 16 + l15] = f2bf(p);
      }
#pragma unroll
      for (int off = 1; off < 16; off <<= 1)
        rs += __shfl_xor(rs, off, 32);
      l_i[r] = l_i[r] * alpha[r] + rs;
    }
#pragma unroll
    for (int dn = 0; dn < 4; ++dn)
#pragma unroll
      for (int r = 0; r < 8; ++r) o[dn][r] *= alpha[r];
    __syncthreads();  // make Ps visible across lanes (+ order dscnt)

    // O += P * V  (A = P from Ps rows; B = V from transposed Vst rows)
    FragB pa[2];
#pragma unroll
    for (int kk = 0; kk < 2; ++kk) {
      const unsigned short* src = &Ps[wav][l15][kk * 32 + grp * 8];
      pa[kk].q[0] = *(const uint4*)src;
      pa[kk].q[1] = *(const uint4*)(src + 16);
    }
#pragma unroll
    for (int dn = 0; dn < 4; ++dn)
#pragma unroll
      for (int kk = 0; kk < 2; ++kk) {
        FragB vb;
        const unsigned short* src = &Vst[dn * 16 + l15][kk * 32 + grp * 8];
        vb.q[0] = *(const uint4*)src;
        vb.q[1] = *(const uint4*)(src + 16);
        o[dn] = __builtin_amdgcn_wmma_f32_16x16x32_bf16(
            false, pa[kk].v, false, vb.v, (short)0, o[dn], false, false);
      }
  }

  // normalize and store bf16
#pragma unroll
  for (int dn = 0; dn < 4; ++dn)
#pragma unroll
    for (int r = 0; r < 8; ++r) {
      int qi = q0 + wav * 16 + grp * 8 + r;
      float val = o[dn][r] / l_i[r];
      vec[((size_t)qi * 8 + b) * 1024 + h * 64 + dn * 16 + l15] = f2bf(val);
    }
}

// ---------------------------------------------------------------------------
// Row LayerNorm over 1024 columns, one block per row.
// ---------------------------------------------------------------------------
__global__ __launch_bounds__(256) void layernorm_rows(
    const float* __restrict__ x, const float* __restrict__ gamma,
    const float* __restrict__ beta, float* __restrict__ out)
{
  __shared__ float red[8];
  const int row = blockIdx.x;
  const float* xr = x + (size_t)row * 1024;
  float v[4];
  float s = 0.f;
#pragma unroll
  for (int i = 0; i < 4; ++i) { v[i] = xr[threadIdx.x + i * 256]; s += v[i]; }
#pragma unroll
  for (int off = 1; off < 32; off <<= 1) s += __shfl_xor(s, off, 32);
  if ((threadIdx.x & 31) == 0) red[threadIdx.x >> 5] = s;
  __syncthreads();
  float tot = 0.f;
#pragma unroll
  for (int w = 0; w < 8; ++w) tot += red[w];
  float mu = tot * (1.f / 1024.f);
  __syncthreads();
  float s2 = 0.f;
#pragma unroll
  for (int i = 0; i < 4; ++i) { float d = v[i] - mu; s2 += d * d; }
#pragma unroll
  for (int off = 1; off < 32; off <<= 1) s2 += __shfl_xor(s2, off, 32);
  if ((threadIdx.x & 31) == 0) red[threadIdx.x >> 5] = s2;
  __syncthreads();
  float tv = 0.f;
#pragma unroll
  for (int w = 0; w < 8; ++w) tv += red[w];
  float rstd = rsqrtf(tv * (1.f / 1024.f) + 1e-5f);
#pragma unroll
  for (int i = 0; i < 4; ++i) {
    int c = threadIdx.x + i * 256;
    out[(size_t)row * 1024 + c] = (v[i] - mu) * rstd * gamma[c] + beta[c];
  }
}

// ---------------------------------------------------------------------------
extern "C" void kernel_launch(void* const* d_in, const int* in_sizes, int n_in,
                              void* d_out, int out_size, void* d_ws, size_t ws_size,
                              hipStream_t stream) {
  const float* q     = (const float*)d_in[0];
  const float* k     = (const float*)d_in[1];
  const float* v     = (const float*)d_in[2];
  // d_in[3] = attn_mask (causal triu, handled analytically)
  const float* Wq    = (const float*)d_in[4];
  const float* Wk    = (const float*)d_in[5];
  const float* Wv    = (const float*)d_in[6];
  const float* Wo    = (const float*)d_in[7];
  const float* gamma = (const float*)d_in[8];
  const float* beta  = (const float*)d_in[9];

  char* ws = (char*)d_ws;
  unsigned short* hq  = (unsigned short*)(ws);                          // 16 MB
  unsigned short* hk  = (unsigned short*)(ws + ((size_t)16 << 20));     // 16 MB
  unsigned short* hv  = (unsigned short*)(ws + ((size_t)32 << 20));     // 16 MB
  unsigned short* vec = (unsigned short*)(ws + ((size_t)48 << 20));     // 16 MB
  float*          att = (float*)        (ws + ((size_t)64 << 20));      // 32 MB

  const int M = 1024 * 8, N = 1024, K = 1024;
  dim3 gemm_grid(N / 128, M / 128);  // (8, 64)

  gemm_bf16_wmma<true,  true ><<<gemm_grid, 256, 0, stream>>>(q,   Wq, hq,  M, N, K);
  gemm_bf16_wmma<true,  true ><<<gemm_grid, 256, 0, stream>>>(k,   Wk, hk,  M, N, K);
  gemm_bf16_wmma<true,  true ><<<gemm_grid, 256, 0, stream>>>(v,   Wv, hv,  M, N, K);
  flash_attn_wmma<<<dim3(1024 / 64, 8 * 16), 128, 0, stream>>>(hq, hk, hv, vec);
  gemm_bf16_wmma<false, false><<<gemm_grid, 256, 0, stream>>>(vec, Wo, att, M, N, K);
  layernorm_rows<<<M, 256, 0, stream>>>(att, gamma, beta, (float*)d_out);
}
